// GCNNet_41120016892600
// MI455X (gfx1250) — matrix operand order repaired and checked by reading, not compiled, not run on hardware
//
#include <hip/hip_runtime.h>
#include <hip/hip_bf16.h>
#include <math.h>

// ---------------------------------------------------------------------------
// GCNNet forward for MI455X (gfx1250, wave32, WMMA).
//   conv: t = X @ W (WMMA f16->f32, weights staged in LDS in WMMA layout),
//         agg = scatter_add(norm * t[src]) + selfloop, x_l = relu(agg + b)
//   head: fc1(concat(x1,x2,x3)) -> fc2a -> segment_sum -> fc2b -> fc3 -> log_softmax
// ---------------------------------------------------------------------------

typedef __attribute__((ext_vector_type(16))) _Float16 v16h;
typedef __attribute__((ext_vector_type(8)))  float    v8f;

#define NNODES 100000
#define NGRAPH 512

// ---- WMMA tile helpers (wave32, 16x16 tiles, K-chunks of 32) --------------
// A is MxK row-major f32; per-lane layout for V_WMMA_F32_16X16X32_F16:
//   lanes 0-15  : row M=lane,    K = {kb+0..7,  kb+16..23}
//   lanes 16-31 : row M=lane-16, K = {kb+8..15, kb+24..31}
__device__ __forceinline__ v16h load_a_tile_f32(const float* __restrict__ A, int lda,
                                                int row_base, int kb, int lane) {
    int m  = lane & 15;
    int ko = (lane & 16) ? 8 : 0;
    const float* p = A + (size_t)(row_base + m) * lda + kb + ko;
    v16h a;
#pragma unroll
    for (int j = 0; j < 8; ++j) a[j] = (_Float16)p[j];
#pragma unroll
    for (int j = 0; j < 8; ++j) a[8 + j] = (_Float16)p[16 + j];
    return a;
}

// Stage B (KxN row-major f32) into LDS as f16, pre-swizzled into the WMMA
// B-operand per-lane layout:
//   lanes 0-15  : col N=lane,    K = {kb+0..7,  kb+16..23}
//   lanes 16-31 : col N=lane-16, K = {kb+8..15, kb+24..31}
// LDS index: (((kchunk * (NC/16) + ntile) * 32 + lane) * 16 + j)
template <int KDIM, int NC>
__device__ __forceinline__ void stage_b_lds(_Float16* lds, const float* __restrict__ B) {
    for (int idx = threadIdx.x; idx < KDIM * NC; idx += blockDim.x) {
        int k = idx / NC, n = idx - k * NC;
        int chunk = k >> 5, kr = k & 31;
        int ntile = n >> 4, nl = n & 15;
        int lane = nl + ((kr & 8) << 1);
        int j    = (kr & 7) | ((kr & 16) >> 1);
        lds[(((chunk * (NC / 16)) + ntile) * 32 + lane) * 16 + j] = (_Float16)B[idx];
    }
}

template <int NC>
__device__ __forceinline__ v16h load_b_lds(const _Float16* lds, int kb, int nb, int lane) {
    int chunk = kb >> 5, ntile = nb >> 4;
    return *(const v16h*)(lds + (size_t)(((chunk * (NC / 16)) + ntile) * 32 + lane) * 16);
}

__device__ __forceinline__ v8f wmma_f16(v16h a, v16h b, v8f c) {
    return __builtin_amdgcn_wmma_f32_16x16x32_f16(false, a, false, b, (short)0, c,
                                                  false, false);
}

// C/D layout: lane l -> col N=(l&15); VGPR r -> row M = r + (l<16 ? 0 : 8)
__device__ __forceinline__ void store_c_tile(float* __restrict__ C, int ldc,
                                             int row_base, int nb, int lane, v8f c,
                                             const float* __restrict__ bias, bool relu) {
    int n  = lane & 15;
    int mo = (lane & 16) ? 8 : 0;
    float bv = bias ? bias[nb + n] : 0.0f;
#pragma unroll
    for (int r = 0; r < 8; ++r) {
        float v = c[r] + bv;
        if (relu) v = fmaxf(v, 0.0f);
        C[(size_t)(row_base + mo + r) * ldc + nb + n] = v;
    }
}

// ---- Generic WMMA GEMM:  C[nrows x NC] = A[nrows x KDIM] @ B[KDIM x NC] ---
// launch: grid = nrows/16 blocks, block = 32 * (NC/16) threads
template <int KDIM, int NC, bool RELU, bool BIAS>
__global__ void gemm_wmma_kernel(const float* __restrict__ A, const float* __restrict__ B,
                                 const float* __restrict__ bias, float* __restrict__ C) {
    __shared__ _Float16 sB[KDIM * NC];
    stage_b_lds<KDIM, NC>(sB, B);
    __syncthreads();

    int lane     = threadIdx.x & 31;
    int nb       = (threadIdx.x >> 5) * 16;   // column tile per wave
    int row_base = blockIdx.x * 16;
    v8f c = {};
#pragma unroll
    for (int kb = 0; kb < KDIM; kb += 32) {
        v16h a = load_a_tile_f32(A, KDIM, row_base, kb, lane);
        v16h b = load_b_lds<NC>(sB, kb, nb, lane);
        c = wmma_f16(a, b, c);
    }
    store_c_tile(C, NC, row_base, nb, lane, c, BIAS ? bias : nullptr, RELU);
}

// ---- fc1 over virtual concat [x1|x2|x3] (K=96, NC=64) ---------------------
__global__ void gemm_concat3_kernel(const float* __restrict__ x1, const float* __restrict__ x2,
                                    const float* __restrict__ x3, const float* __restrict__ B,
                                    const float* __restrict__ bias, float* __restrict__ C) {
    const int NC = 64;
    __shared__ _Float16 sB[96 * NC];
    stage_b_lds<96, NC>(sB, B);
    __syncthreads();

    int lane     = threadIdx.x & 31;
    int nb       = (threadIdx.x >> 5) * 16;
    int row_base = blockIdx.x * 16;
    const float* Xs[3] = {x1, x2, x3};
    v8f c = {};
#pragma unroll
    for (int ch = 0; ch < 3; ++ch) {
        v16h a = load_a_tile_f32(Xs[ch], 32, row_base, 0, lane);
        v16h b = load_b_lds<NC>(sB, ch * 32, nb, lane);
        c = wmma_f16(a, b, c);
    }
    store_c_tile(C, NC, row_base, nb, lane, c, bias, true);
}

// ---- degree / normalization ----------------------------------------------
__global__ void deg_init_kernel(float* deg, int n) {
    int i = blockIdx.x * blockDim.x + threadIdx.x;
    if (i < n) deg[i] = 1.0f;   // self-loop
}
__global__ void deg_accum_kernel(const int* __restrict__ dst, float* deg, int ne) {
    int e = blockIdx.x * blockDim.x + threadIdx.x;
    if (e < ne) atomicAdd(&deg[dst[e]], 1.0f);
}
__global__ void deg_rsqrt_kernel(float* deg, int n) {
    int i = blockIdx.x * blockDim.x + threadIdx.x;
    if (i < n) deg[i] = rsqrtf(deg[i]);   // in place: deg -> dinv
}

// ---- aggregation ----------------------------------------------------------
// agg[i][f] = t[i][f] * dinv[i]^2   (self-loop term; also zero-inits agg)
__global__ void agg_init_kernel(const float* __restrict__ t, const float* __restrict__ dinv,
                                float* __restrict__ agg, int n) {
    int idx = blockIdx.x * blockDim.x + threadIdx.x;
    if (idx < n * 32) {
        int node = idx >> 5;
        float di = dinv[node];
        agg[idx] = t[idx] * di * di;
    }
}

// one wave per edge per iteration; lane = feature; look-ahead prefetch of the
// next random source row (global_prefetch_b8)
__global__ void edge_scatter_kernel(const int* __restrict__ src, const int* __restrict__ dst,
                                    const float* __restrict__ dinv, const float* __restrict__ t,
                                    float* __restrict__ agg, int ne) {
    int lane   = threadIdx.x & 31;
    int wave   = (blockIdx.x * blockDim.x + threadIdx.x) >> 5;
    int nwaves = (gridDim.x * blockDim.x) >> 5;
    for (int e = wave; e < ne; e += nwaves) {
        int s = src[e];
        int d = dst[e];
        int en = e + nwaves;
        if (en < ne) {
            int sn = src[en];
            __builtin_prefetch(t + (size_t)sn * 32 + lane, 0, 0);
        }
        float norm = dinv[s] * dinv[d];
        atomicAdd(&agg[(size_t)d * 32 + lane], t[(size_t)s * 32 + lane] * norm);
    }
}

__global__ void bias_relu_kernel(const float* __restrict__ agg, const float* __restrict__ b,
                                 float* __restrict__ out, int total) {
    int idx = blockIdx.x * blockDim.x + threadIdx.x;
    if (idx < total) out[idx] = fmaxf(agg[idx] + b[idx & 31], 0.0f);
}

// ---- pooling (segment_sum over batch) ------------------------------------
__global__ void zero_kernel(float* p, int n) {
    int i = blockIdx.x * blockDim.x + threadIdx.x;
    if (i < n) p[i] = 0.0f;
}
__global__ void pool_kernel(const float* __restrict__ g2, const int* __restrict__ batch,
                            float* __restrict__ pooled, int n) {
    int node = blockIdx.x * 2 + (threadIdx.x >> 7);
    int f    = threadIdx.x & 127;
    if (node < n)
        atomicAdd(&pooled[(size_t)batch[node] * 128 + f], g2[(size_t)node * 128 + f]);
}

// ---- head: fc3 + log_softmax (OUT = 2) ------------------------------------
__global__ void head_kernel(const float* __restrict__ p1, const float* __restrict__ w,
                            const float* __restrict__ b, float* __restrict__ out) {
    int r = blockIdx.x * blockDim.x + threadIdx.x;
    if (r >= NGRAPH) return;
    const float* row = p1 + (size_t)r * 64;
    float z0 = b[0], z1 = b[1];
#pragma unroll 8
    for (int k = 0; k < 64; ++k) {
        float v = row[k];
        z0 += v * w[k * 2 + 0];
        z1 += v * w[k * 2 + 1];
    }
    float m   = fmaxf(z0, z1);
    float lse = m + logf(expf(z0 - m) + expf(z1 - m));
    out[r * 2 + 0] = z0 - lse;
    out[r * 2 + 1] = z1 - lse;
}

// ---------------------------------------------------------------------------
extern "C" void kernel_launch(void* const* d_in, const int* in_sizes, int n_in,
                              void* d_out, int out_size, void* d_ws, size_t ws_size,
                              hipStream_t stream) {
    const float* x      = (const float*)d_in[0];
    const int*   eidx   = (const int*)d_in[1];
    const int*   batch  = (const int*)d_in[2];
    const float* W1     = (const float*)d_in[3];
    const float* b1     = (const float*)d_in[4];
    const float* W2     = (const float*)d_in[5];
    const float* b2     = (const float*)d_in[6];
    const float* W3     = (const float*)d_in[7];
    const float* b3     = (const float*)d_in[8];
    const float* fc1_w  = (const float*)d_in[9];
    const float* fc1_b  = (const float*)d_in[10];
    const float* fc2a_w = (const float*)d_in[11];
    const float* fc2a_b = (const float*)d_in[12];
    const float* fc2b_w = (const float*)d_in[13];
    const float* fc2b_b = (const float*)d_in[14];
    const float* fc3_w  = (const float*)d_in[15];
    const float* fc3_b  = (const float*)d_in[16];

    const int N = in_sizes[0] / 128;   // 100000 (multiple of 16)
    const int E = in_sizes[1] / 2;     // 3200000
    const int* src = eidx;
    const int* dst = eidx + E;

    // workspace layout (floats)
    float* ws     = (float*)d_ws;
    float* dinv   = ws;                            // N
    float* t      = dinv + N;                      // N*32
    float* agg    = t + (size_t)N * 32;            // N*32
    float* x1     = agg + (size_t)N * 32;          // N*32
    float* x2     = x1 + (size_t)N * 32;           // N*32
    float* x3     = x2 + (size_t)N * 32;           // N*32
    float* g1     = x3 + (size_t)N * 32;           // N*64
    float* g2     = g1 + (size_t)N * 64;           // N*128
    float* pooled = g2 + (size_t)N * 128;          // 512*128
    float* p1     = pooled + (size_t)NGRAPH * 128; // 512*64

    const int rowTiles = N / 16;     // 6250
    const int TPB = 256;

    // --- degrees -> dinv ---
    deg_init_kernel<<<(N + TPB - 1) / TPB, TPB, 0, stream>>>(dinv, N);
    deg_accum_kernel<<<(E + TPB - 1) / TPB, TPB, 0, stream>>>(dst, dinv, E);
    deg_rsqrt_kernel<<<(N + TPB - 1) / TPB, TPB, 0, stream>>>(dinv, N);

    const int scatterBlocks = 8192;  // 65536 waves, grid-stride over edges

    // --- GCN layer 1 (K=128 -> 32) ---
    gemm_wmma_kernel<128, 32, false, false><<<rowTiles, 64, 0, stream>>>(x, W1, nullptr, t);
    agg_init_kernel<<<(N * 32 + TPB - 1) / TPB, TPB, 0, stream>>>(t, dinv, agg, N);
    edge_scatter_kernel<<<scatterBlocks, TPB, 0, stream>>>(src, dst, dinv, t, agg, E);
    bias_relu_kernel<<<(N * 32 + TPB - 1) / TPB, TPB, 0, stream>>>(agg, b1, x1, N * 32);

    // --- GCN layer 2 (32 -> 32) ---
    gemm_wmma_kernel<32, 32, false, false><<<rowTiles, 64, 0, stream>>>(x1, W2, nullptr, t);
    agg_init_kernel<<<(N * 32 + TPB - 1) / TPB, TPB, 0, stream>>>(t, dinv, agg, N);
    edge_scatter_kernel<<<scatterBlocks, TPB, 0, stream>>>(src, dst, dinv, t, agg, E);
    bias_relu_kernel<<<(N * 32 + TPB - 1) / TPB, TPB, 0, stream>>>(agg, b2, x2, N * 32);

    // --- GCN layer 3 (32 -> 32) ---
    gemm_wmma_kernel<32, 32, false, false><<<rowTiles, 64, 0, stream>>>(x2, W3, nullptr, t);
    agg_init_kernel<<<(N * 32 + TPB - 1) / TPB, TPB, 0, stream>>>(t, dinv, agg, N);
    edge_scatter_kernel<<<scatterBlocks, TPB, 0, stream>>>(src, dst, dinv, t, agg, E);
    bias_relu_kernel<<<(N * 32 + TPB - 1) / TPB, TPB, 0, stream>>>(agg, b3, x3, N * 32);

    // --- fc1 over concat(x1,x2,x3): K=96 -> 64, relu ---
    gemm_concat3_kernel<<<rowTiles, 128, 0, stream>>>(x1, x2, x3, fc1_w, fc1_b, g1);

    // --- fc2a: K=64 -> 128, relu ---
    gemm_wmma_kernel<64, 128, true, true><<<rowTiles, 256, 0, stream>>>(g1, fc2a_w, fc2a_b, g2);

    // --- global_add_pool ---
    zero_kernel<<<(NGRAPH * 128 + TPB - 1) / TPB, TPB, 0, stream>>>(pooled, NGRAPH * 128);
    pool_kernel<<<(N + 1) / 2, 256, 0, stream>>>(g2, batch, pooled, N);

    // --- fc2b: 512 x (128 -> 64), relu ---
    gemm_wmma_kernel<128, 64, true, true><<<NGRAPH / 16, 128, 0, stream>>>(pooled, fc2b_w,
                                                                           fc2b_b, p1);

    // --- fc3 + log_softmax ---
    head_kernel<<<(NGRAPH + TPB - 1) / TPB, TPB, 0, stream>>>(p1, fc3_w, fc3_b, (float*)d_out);
}
